// RWKV_TimeMix_x051a_28879360099039
// MI455X (gfx1250) — matrix operand order repaired and checked by reading, not compiled, not run on hardware
//
#include <hip/hip_runtime.h>

// ---------------------------------------------------------------------------
// RWKV-5 TimeMix (B=4, T=2048, D=1024, H=16, N=64, Q=256) for gfx1250 (CDNA5)
// - all GEMM-shaped math on v_wmma_f32_16x16x32_bf16 (wave32 WMMA)
// - LDS tile staging via global_load_async_to_lds_b128 (ASYNCcnt engine),
//   double-buffered so the DMA overlaps the WMMA burst
// ---------------------------------------------------------------------------

typedef __attribute__((ext_vector_type(16))) __bf16 v16bf;
typedef __attribute__((ext_vector_type(8)))  __bf16 bf16x8;
typedef __attribute__((ext_vector_type(8)))  float  v8f;

#define DD   1024
#define HH   16
#define NN   64
#define BB   4
#define TT   2048
#define QQ   256
#define MTOT 8192   // B*T

static __device__ __forceinline__ v8f wmma_bf16(v16bf a, v16bf b, v8f c) {
  return __builtin_amdgcn_wmma_f32_16x16x32_bf16(false, a, false, b, (short)0, c,
                                                 false, false);
}

// Fast sigmoid (bf16-accuracy): one v_rcp_f32 instead of IEEE divide.
static __device__ __forceinline__ float fast_silu(float v) {
  return v * __builtin_amdgcn_rcpf(1.f + __expf(-v));
}

// Async DMA: global memory -> LDS, 16 bytes per lane, tracked by ASYNCcnt.
// ldsoff is the wave-relative LDS byte address (low 32 bits of a generic
// pointer into __shared__ memory -- the LDS aperture keeps the offset there).
static __device__ __forceinline__ void async_b128(unsigned ldsoff,
                                                  const __bf16* g) {
  asm volatile("global_load_async_to_lds_b128 %0, %1, off"
               :: "v"(ldsoff), "v"(g) : "memory");
}
static __device__ __forceinline__ void wait_async0() {
  asm volatile("s_wait_asynccnt 0" ::: "memory");
}
static __device__ __forceinline__ unsigned lds_off(const void* p) {
  return (unsigned)(unsigned long long)p;
}

// A fragment (16x32 bf16, MxK): lane L -> M = mbase + (L&15);
// packed halves: K = klo..klo+7 and klo+16..klo+23 with klo = kbase + (L>>4)*8.
static __device__ __forceinline__ v16bf frag_a(const __bf16* p, int stride,
                                               int mbase, int kbase, int lane) {
  const int m   = mbase + (lane & 15);
  const int klo = kbase + ((lane >> 4) << 3);
  union { v16bf v; bf16x8 h[2]; } u;
  u.h[0] = *(const bf16x8*)(p + (size_t)m * stride + klo);
  u.h[1] = *(const bf16x8*)(p + (size_t)m * stride + klo + 16);
  return u.v;
}

// B fragment (32x16 bf16, KxN) sourced from a row-major [N][K] buffer
// (B[k][n] = buf[n][k]): lane L -> N = nbase + (L&15),
// packed halves: K = klo..klo+7 and klo+16..klo+23 with klo = kbase + (L>>4)*8.
static __device__ __forceinline__ v16bf frag_b(const __bf16* p, int stride,
                                               int nbase, int kbase, int lane) {
  const int n = nbase + (lane & 15);
  const int k = kbase + ((lane >> 4) << 3);
  union { v16bf v; bf16x8 h[2]; } u;
  u.h[0] = *(const bf16x8*)(p + (size_t)n * stride + k);
  u.h[1] = *(const bf16x8*)(p + (size_t)n * stride + k + 16);
  return u.v;
}

// ---------------------------------------------------------------------------
// Kernel: f32 -> bf16 convert (weights)
// ---------------------------------------------------------------------------
__global__ void f2bf(const float* __restrict__ src, __bf16* __restrict__ dst, int n) {
  int i = blockIdx.x * 256 + threadIdx.x;
  if (i < n) dst[i] = (__bf16)src[i];
}

// ---------------------------------------------------------------------------
// Kernel: time-shift mix -> xk, xv, xr, xg (bf16)
// ---------------------------------------------------------------------------
__global__ void mix_kernel(const float* __restrict__ x,
                           const float* __restrict__ mk, const float* __restrict__ mv,
                           const float* __restrict__ mr, const float* __restrict__ mg,
                           __bf16* __restrict__ xk, __bf16* __restrict__ xv,
                           __bf16* __restrict__ xr, __bf16* __restrict__ xg) {
  size_t i = (size_t)blockIdx.x * 256 + threadIdx.x;
  int d = (int)(i & (DD - 1));
  int t = (int)((i >> 10) & (TT - 1));
  float xi = x[i];
  float xs = (t == 0) ? 0.f : x[i - DD];
  float dx = xs - xi;
  xk[i] = (__bf16)(xi + dx * mk[d]);
  xv[i] = (__bf16)(xi + dx * mv[d]);
  xr[i] = (__bf16)(xi + dx * mr[d]);
  xg[i] = (__bf16)(xi + dx * mg[d]);
}

// ---------------------------------------------------------------------------
// Kernel: C[M,1024] = A[M,1024](bf16) @ W[1024,1024](bf16)^T   (NT GEMM)
// Block tile 128x128, K-step 64, 8 waves -> each wave 32x64 (16 WMMA/stage).
// Double-buffered LDS; next stage is DMA'd by the async engine while the
// current stage's WMMAs run.
// mode 0: store bf16; mode 1: SiLU then bf16; mode 2: store f32.
// ---------------------------------------------------------------------------
__global__ void __launch_bounds__(256)
gemm_nt(const __bf16* __restrict__ A, const __bf16* __restrict__ W,
        __bf16* __restrict__ outB, float* __restrict__ outF, int mode) {
  __shared__ __bf16 As[2][128 * 72];
  __shared__ __bf16 Bs[2][128 * 72];
  const int lane = threadIdx.x & 31, wave = threadIdx.x >> 5;
  const int wm = wave >> 1, wn = wave & 1;
  const int m0 = blockIdx.y * 128, n0 = blockIdx.x * 128;

  v8f acc[2][4];
#pragma unroll
  for (int i = 0; i < 2; i++)
#pragma unroll
    for (int j = 0; j < 4; j++)
      acc[i][j] = (v8f){0.f, 0.f, 0.f, 0.f, 0.f, 0.f, 0.f, 0.f};

  const int lrow = threadIdx.x >> 1;           // 2 threads per row
  const int lcs  = (threadIdx.x & 1) * 32;     // each covers 32 contiguous bf16

  // stage(buf, k0): async-DMA the 128x64 A and W tiles into LDS buffer `buf`
  auto stage = [&](int buf, int k0) {
    unsigned la = lds_off(&As[buf][lrow * 72 + lcs]);
    unsigned lb = lds_off(&Bs[buf][lrow * 72 + lcs]);
    const __bf16* ga = A + (size_t)(m0 + lrow) * DD + k0 + lcs;
    const __bf16* gw = W + (size_t)(n0 + lrow) * DD + k0 + lcs;
#pragma unroll
    for (int cch = 0; cch < 4; cch++) {
      async_b128(la + cch * 16, ga + cch * 8);
      async_b128(lb + cch * 16, gw + cch * 8);
    }
  };

  stage(0, 0);
  wait_async0();
  __syncthreads();

  for (int k0 = 0; k0 < DD; k0 += 64) {
    const int cur = (k0 >> 6) & 1;
    if (k0 + 64 < DD) stage(cur ^ 1, k0 + 64);  // DMA next tile during math

    const __bf16* Ac = As[cur];
    const __bf16* Bc = Bs[cur];
#pragma unroll
    for (int ks = 0; ks < 64; ks += 32) {
      v16bf af[2], bfr[4];
#pragma unroll
      for (int i = 0; i < 2; i++) af[i] = frag_a(Ac, 72, wm * 32 + i * 16, ks, lane);
#pragma unroll
      for (int j = 0; j < 4; j++) bfr[j] = frag_b(Bc, 72, wn * 64 + j * 16, ks, lane);
#pragma unroll
      for (int i = 0; i < 2; i++)
#pragma unroll
        for (int j = 0; j < 4; j++)
          acc[i][j] = wmma_bf16(af[i], bfr[j], acc[i][j]);
    }
    wait_async0();      // this wave's next-stage DMA landed
    __syncthreads();    // everyone done with `cur`, next stage visible
  }

#pragma unroll
  for (int i = 0; i < 2; i++)
#pragma unroll
    for (int j = 0; j < 4; j++) {
      int rbase = m0 + wm * 32 + i * 16 + ((lane >> 4) << 3);
      int c     = n0 + wn * 64 + j * 16 + (lane & 15);
#pragma unroll
      for (int e = 0; e < 8; e++) {
        float v = acc[i][j][e];
        size_t idx = (size_t)(rbase + e) * DD + c;
        if (mode == 0)      outB[idx] = (__bf16)v;
        else if (mode == 1) outB[idx] = (__bf16)fast_silu(v);
        else                outF[idx] = v;
      }
    }
}

// ---------------------------------------------------------------------------
// Kernel: chunked linear attention. One workgroup per (b,h). 16 waves (512).
// Dynamic LDS layout (bytes):
//   kt  [256][72] bf16    k rows (B operand of S = r@k), async-DMA staged
//   vT  [ 64][264] bf16   v transposed (B operand of S@v and state update)
//   kTs [ 64][264] bf16   k^T scaled by w^(Q-1-q) (A operand of state update)
//   stF [ 64][65] f32     running state
//   stB [ 64][72] bf16    bf16 mirror of state (B operand of r@state)
//   wpow[260] f32         w^0 .. w^256
//   sS  [16][16][264] bf16 per-wave masked S staging (A operand of S@v)
// Total = 266512 bytes < 320 KB WGP limit (one WG owns the WGP).
// ---------------------------------------------------------------------------
__global__ void __launch_bounds__(512)
rwkv_attn(const __bf16* __restrict__ rb, const __bf16* __restrict__ kb,
          const __bf16* __restrict__ vb, const float* __restrict__ decay,
          const float* __restrict__ faaaa, float* __restrict__ yF) {
  extern __shared__ char smem[];
  __bf16* kt   = (__bf16*)smem;               // 256*72
  __bf16* vT   = kt + 256 * 72;               // 64*264
  __bf16* kTs  = vT + 64 * 264;               // 64*264
  float*  stF  = (float*)(kTs + 64 * 264);    // 64*65
  __bf16* stB  = (__bf16*)(stF + 64 * 65);    // 64*72
  float*  wpow = (float*)(stB + 64 * 72);     // 260
  __bf16* sS   = (__bf16*)(wpow + 260);       // 16*16*264

  const int b = blockIdx.x >> 4, h = blockIdx.x & 15;
  const int lane = threadIdx.x & 31, wave = threadIdx.x >> 5;   // wave 0..15
  const int h64 = h * 64;
  const float w = __expf(-__expf(decay[h]));
  const float u = faaaa[h];

  for (int i = threadIdx.x; i < 64 * 65; i += 512) stF[i] = 0.f;
  for (int i = threadIdx.x; i < 64 * 72; i += 512) stB[i] = (__bf16)0.f;
  for (int i = threadIdx.x; i < 257; i += 512) wpow[i] = __powf(w, (float)i);

  __bf16* sSw = sS + wave * 16 * 264;
  const int rowbase = wave * 16;               // one 16-row block per wave

  for (int c = 0; c < TT / QQ; c++) {
    const size_t bT0 = (size_t)b * TT + (size_t)c * QQ;
    __syncthreads();
    // kt tile: straight row copy -> async DMA engine (16B chunks)
    for (int i = threadIdx.x; i < 2048; i += 512) {
      int q  = i >> 3;
      int n8 = (i & 7) * 8;
      async_b128(lds_off(kt + q * 72 + n8),
                 kb + (bT0 + q) * DD + h64 + n8);
    }
    // transposed / decay-scaled tiles: scalar path
    for (int i = threadIdx.x; i < 64 * 256; i += 512) {
      int n = i & 63, q = i >> 6;
      __bf16 kv = kb[(bT0 + q) * DD + h64 + n];
      __bf16 vv = vb[(bT0 + q) * DD + h64 + n];
      vT[n * 264 + q]  = vv;
      kTs[n * 264 + q] = (__bf16)((float)kv * wpow[255 - q]);
    }
    wait_async0();
    __syncthreads();

    // ---- per-wave 16-row block -------------------------------------------
    {
      const __bf16* rrow = rb + bT0 * DD + h64;
      v16bf ra0 = frag_a(rrow, DD, rowbase, 0, lane);
      v16bf ra1 = frag_a(rrow, DD, rowbase, 32, lane);

      // S = r @ k  (16 x 256), mask with decay/u, stage as bf16
      for (int jt = 0; jt < 16; jt++) {
        v8f s = (v8f){0.f, 0.f, 0.f, 0.f, 0.f, 0.f, 0.f, 0.f};
        s = wmma_bf16(ra0, frag_b(kt, 72, jt * 16, 0, lane), s);
        s = wmma_bf16(ra1, frag_b(kt, 72, jt * 16, 32, lane), s);
#pragma unroll
        for (int e = 0; e < 8; e++) {
          int rloc = e + ((lane >> 4) << 3);
          int qi = rowbase + rloc;
          int qj = jt * 16 + (lane & 15);
          float f = (qj < qi) ? wpow[qi - qj - 1] : ((qj == qi) ? u : 0.f);
          sSw[rloc * 264 + qj] = (__bf16)(s[e] * f);
        }
      }

      // Y = (r @ state) * w^qi  +  Smasked @ v
      v8f acc[4];
#pragma unroll
      for (int jt = 0; jt < 4; jt++)
        acc[jt] = (v8f){0.f, 0.f, 0.f, 0.f, 0.f, 0.f, 0.f, 0.f};
#pragma unroll
      for (int jt = 0; jt < 4; jt++) {
        acc[jt] = wmma_bf16(ra0, frag_b(stB, 72, jt * 16, 0, lane), acc[jt]);
        acc[jt] = wmma_bf16(ra1, frag_b(stB, 72, jt * 16, 32, lane), acc[jt]);
      }
#pragma unroll
      for (int jt = 0; jt < 4; jt++)
#pragma unroll
        for (int e = 0; e < 8; e++)
          acc[jt][e] *= wpow[rowbase + e + ((lane >> 4) << 3)];

      for (int ks = 0; ks < 8; ks++) {
        v16bf sa = frag_a(sSw, 264, 0, ks * 32, lane);
#pragma unroll
        for (int jt = 0; jt < 4; jt++)
          acc[jt] = wmma_bf16(sa, frag_b(vT, 264, jt * 16, ks * 32, lane), acc[jt]);
      }

#pragma unroll
      for (int jt = 0; jt < 4; jt++)
#pragma unroll
        for (int e = 0; e < 8; e++) {
          int rloc = e + ((lane >> 4) << 3);
          yF[(bT0 + rowbase + rloc) * DD + h64 + jt * 16 + (lane & 15)] = acc[jt][e];
        }
    }

    // ---- state update: state = w^Q * state + (k*wk) @ v -------------------
    // 16 tiles (4x4 over 64x64), exactly one per wave.
    v8f sacc;
    const float wsQ = wpow[256];
    const int rt = wave >> 2, ct = wave & 3;
#pragma unroll
    for (int e = 0; e < 8; e++)
      sacc[e] = wsQ * stF[(rt * 16 + e + ((lane >> 4) << 3)) * 65 +
                          ct * 16 + (lane & 15)];
    for (int ks = 0; ks < 8; ks++) {
      v16bf ka = frag_a(kTs, 264, rt * 16, ks * 32, lane);
      sacc = wmma_bf16(ka, frag_b(vT, 264, ct * 16, ks * 32, lane), sacc);
    }
    __syncthreads();
#pragma unroll
    for (int e = 0; e < 8; e++) {
      int rr = rt * 16 + e + ((lane >> 4) << 3);
      int cc = ct * 16 + (lane & 15);
      stF[rr * 65 + cc] = sacc[e];
      stB[rr * 72 + cc] = (__bf16)sacc[e];
    }
  }
}

// ---------------------------------------------------------------------------
// Kernel: per-(b,t,h) group norm over N=64, LN affine, * silu(g) -> bf16
// One wave per row (wave32: 2 elements per lane).
// ---------------------------------------------------------------------------
__global__ void __launch_bounds__(256)
groupnorm_gate(const float* __restrict__ yF, const __bf16* __restrict__ gb,
               const float* __restrict__ lng, const float* __restrict__ lnb,
               __bf16* __restrict__ a2) {
  const int wave = threadIdx.x >> 5, lane = threadIdx.x & 31;
  const long row = (long)blockIdx.x * 8 + wave;   // over B*T*H
  const long bt  = row >> 4;
  const int  h   = (int)(row & 15);
  const float* yp = yF + bt * DD + h * 64;
  float v0 = yp[lane], v1 = yp[lane + 32];
  float s = v0 + v1, q = v0 * v0 + v1 * v1;
#pragma unroll
  for (int off = 16; off; off >>= 1) {
    s += __shfl_xor(s, off, 32);
    q += __shfl_xor(q, off, 32);
  }
  const float mu  = s * (1.f / 64.f);
  const float var = q * (1.f / 64.f) - mu * mu;
  const float inv = rsqrtf(var + 6.4e-4f);   // EPS = 1e-5 * 64
  const int d0 = h * 64 + lane, d1 = d0 + 32;
  float o0 = ((v0 - mu) * inv * lng[d0] + lnb[d0]) * (float)gb[bt * DD + d0];
  float o1 = ((v1 - mu) * inv * lng[d1] + lnb[d1]) * (float)gb[bt * DD + d1];
  a2[bt * DD + d0] = (__bf16)o0;
  a2[bt * DD + d1] = (__bf16)o1;
}

// ---------------------------------------------------------------------------
// Host launcher
// ---------------------------------------------------------------------------
static constexpr size_t MB = 1024ull * 1024ull;
static constexpr size_t O_XK = 0;                 // mix outputs (bf16, 16 MiB each)
static constexpr size_t O_XV = O_XK + 16 * MB;
static constexpr size_t O_XR = O_XV + 16 * MB;
static constexpr size_t O_XG = O_XR + 16 * MB;
static constexpr size_t O_WR = O_XG + 16 * MB;    // bf16 weights (2 MiB each)
static constexpr size_t O_WK = O_WR + 2 * MB;
static constexpr size_t O_WV = O_WK + 2 * MB;
static constexpr size_t O_WG = O_WV + 2 * MB;
static constexpr size_t O_WO = O_WG + 2 * MB;
static constexpr size_t O_R  = O_WO + 2 * MB;     // r (bf16, 16 MiB)
static constexpr size_t O_Y  = O_R + 16 * MB;     // y (f32, 32 MiB)
// reuse: k -> O_XR, v -> O_XK, g -> O_XV, a2 -> O_XG (inputs already consumed)

static constexpr size_t ATT_SMEM =
    (size_t)(256 * 72 + 64 * 264 + 64 * 264 + 64 * 72 + 16 * 16 * 264) * 2 +
    (size_t)(64 * 65 + 260) * 4;                  // 266512 bytes (< 320 KB WGP)

extern "C" void kernel_launch(void* const* d_in, const int* in_sizes, int n_in,
                              void* d_out, int out_size, void* d_ws, size_t ws_size,
                              hipStream_t stream) {
  (void)in_sizes; (void)n_in; (void)out_size; (void)ws_size;
  const float* x     = (const float*)d_in[0];
  const float* w_r   = (const float*)d_in[1];
  const float* w_k   = (const float*)d_in[2];
  const float* w_v   = (const float*)d_in[3];
  const float* w_g   = (const float*)d_in[4];
  const float* w_o   = (const float*)d_in[5];
  const float* mk    = (const float*)d_in[6];
  const float* mv    = (const float*)d_in[7];
  const float* mr    = (const float*)d_in[8];
  const float* mg    = (const float*)d_in[9];
  const float* dec   = (const float*)d_in[10];
  const float* faa   = (const float*)d_in[11];
  const float* lng   = (const float*)d_in[12];
  const float* lnb   = (const float*)d_in[13];
  float* out = (float*)d_out;

  char* ws = (char*)d_ws;
  __bf16* xkb = (__bf16*)(ws + O_XK);
  __bf16* xvb = (__bf16*)(ws + O_XV);
  __bf16* xrb = (__bf16*)(ws + O_XR);
  __bf16* xgb = (__bf16*)(ws + O_XG);
  __bf16* wrb = (__bf16*)(ws + O_WR);
  __bf16* wkb = (__bf16*)(ws + O_WK);
  __bf16* wvb = (__bf16*)(ws + O_WV);
  __bf16* wgb = (__bf16*)(ws + O_WG);
  __bf16* wob = (__bf16*)(ws + O_WO);
  __bf16* rbuf = (__bf16*)(ws + O_R);
  __bf16* kbuf = (__bf16*)(ws + O_XR);   // overwrites consumed xr
  __bf16* vbuf = (__bf16*)(ws + O_XK);   // overwrites consumed xk
  __bf16* gbuf = (__bf16*)(ws + O_XV);   // overwrites consumed xv
  __bf16* a2   = (__bf16*)(ws + O_XG);   // overwrites consumed xg
  float*  yF   = (float*)(ws + O_Y);

  // weights -> bf16
  f2bf<<<4096, 256, 0, stream>>>(w_r, wrb, DD * DD);
  f2bf<<<4096, 256, 0, stream>>>(w_k, wkb, DD * DD);
  f2bf<<<4096, 256, 0, stream>>>(w_v, wvb, DD * DD);
  f2bf<<<4096, 256, 0, stream>>>(w_g, wgb, DD * DD);
  f2bf<<<4096, 256, 0, stream>>>(w_o, wob, DD * DD);

  // time-shift mixing
  mix_kernel<<<MTOT * DD / 256, 256, 0, stream>>>(x, mk, mv, mr, mg,
                                                  xkb, xvb, xrb, xgb);

  // r/k/v/g projections (WMMA GEMMs, async-staged)
  dim3 gg(DD / 128, MTOT / 128);
  gemm_nt<<<gg, 256, 0, stream>>>(xrb, wrb, rbuf, nullptr, 0);
  gemm_nt<<<gg, 256, 0, stream>>>(xkb, wkb, kbuf, nullptr, 0);
  gemm_nt<<<gg, 256, 0, stream>>>(xvb, wvb, vbuf, nullptr, 0);
  gemm_nt<<<gg, 256, 0, stream>>>(xgb, wgb, gbuf, nullptr, 1);  // fused SiLU

  // chunked linear attention (16 waves per (b,h) workgroup)
  rwkv_attn<<<BB * HH, 512, ATT_SMEM, stream>>>(rbuf, kbuf, vbuf, dec, faa, yF);

  // group-norm + gate
  groupnorm_gate<<<BB * TT * HH / 8, 256, 0, stream>>>(yF, gbuf, lng, lnb, a2);

  // output projection
  gemm_nt<<<gg, 256, 0, stream>>>(a2, wob, nullptr, out, 2);
}
